// MultiHeadAttention_22016002359527
// MI455X (gfx1250) — compile-verified, matrix operand order checked by reading, not run on hardware
//
#include <hip/hip_runtime.h>

// Problem constants (from reference)
#define B_   4
#define S_   2048
#define H_   16
#define DH_  64
#define E_   1024
#define F_   3072   // 3*E

typedef __attribute__((ext_vector_type(16))) __bf16 bf16x16;
typedef __attribute__((ext_vector_type(8)))  float  f32x8;

union FragU { bf16x16 v; uint4 q[2]; };

// Load a 16x32 (bf16) A/B fragment: this lane's row/col base pointer plus the two
// 8-element contiguous chunks selected by k0/k1 (ISA 7.12.2 16-bit layout).
__device__ __forceinline__ bf16x16 load_frag(const __bf16* rowp, int k0, int k1) {
    FragU f;
    f.q[0] = *reinterpret_cast<const uint4*>(rowp + k0);
    f.q[1] = *reinterpret_cast<const uint4*>(rowp + k1);
    return f.v;
}

__device__ __forceinline__ f32x8 wmma_bf16(bf16x16 a, bf16x16 b, f32x8 c) {
    // D = A(16x32) x B(32x16) + C, f32 accumulate
    return __builtin_amdgcn_wmma_f32_16x16x32_bf16(false, a, false, b, (short)0, c, false, false);
}

// Butterfly reductions across the 16 lanes of a half-wave (rows of a C-fragment
// live entirely within one 16-lane half on wave32).
__device__ __forceinline__ float rmax16(float v) {
    v = fmaxf(v, __shfl_xor(v, 1, 32));
    v = fmaxf(v, __shfl_xor(v, 2, 32));
    v = fmaxf(v, __shfl_xor(v, 4, 32));
    v = fmaxf(v, __shfl_xor(v, 8, 32));
    return v;
}
__device__ __forceinline__ float rsum16(float v) {
    v += __shfl_xor(v, 1, 32);
    v += __shfl_xor(v, 2, 32);
    v += __shfl_xor(v, 4, 32);
    v += __shfl_xor(v, 8, 32);
    return v;
}

// ---------------------------------------------------------------------------
// Stage 1: f32 -> bf16 conversion (4 elements / thread, 8B stores)
// ---------------------------------------------------------------------------
__global__ __launch_bounds__(256) void cvt_f32_bf16(const float* __restrict__ src,
                                                    unsigned short* __restrict__ dst,
                                                    int n4) {
    int i = blockIdx.x * blockDim.x + threadIdx.x;
    if (i >= n4) return;
    float4 v = reinterpret_cast<const float4*>(src)[i];
    union { __bf16 h[4]; uint2 u2; } p;
    p.h[0] = (__bf16)v.x; p.h[1] = (__bf16)v.y;
    p.h[2] = (__bf16)v.z; p.h[3] = (__bf16)v.w;
    reinterpret_cast<uint2*>(dst)[i] = p.u2;
}

// ---------------------------------------------------------------------------
// Stage 2: QKV GEMM.  C(8192x3072) = X(8192x1024,bf16) * Wqkv^T.
// One wave owns a 32x64 tile (2 row-frags x 4 col-frags -> 8 WMMA / k-step).
// Epilogue scatters bf16 into Q(b,h,s,d), K(b,h,s,d), Vt(b,h,d,s).
// ---------------------------------------------------------------------------
__global__ __launch_bounds__(128) void gemm_qkv(const unsigned short* __restrict__ Xus,
                                                const unsigned short* __restrict__ Wus,
                                                unsigned short* __restrict__ Qus,
                                                unsigned short* __restrict__ Kus,
                                                unsigned short* __restrict__ Vtus) {
    const __bf16* X = reinterpret_cast<const __bf16*>(Xus);
    const __bf16* W = reinterpret_cast<const __bf16*>(Wus);
    __bf16* Qb = reinterpret_cast<__bf16*>(Qus);
    __bf16* Kb = reinterpret_cast<__bf16*>(Kus);
    __bf16* Vt = reinterpret_cast<__bf16*>(Vtus);

    const int lane = threadIdx.x & 31;
    const int wave = threadIdx.x >> 5;
    const int gid  = blockIdx.x * 4 + wave;      // 12288 wave tiles
    const int tn   = gid % 48;                   // 48 col tiles of 64 (N=3072)
    const int tm   = gid / 48;                   // 256 row tiles of 32 (M=8192)
    const int n    = lane & 15;
    const int hlf  = lane >> 4;
    const int rowbase = tm * 32;
    const int colbase = tn * 64;

    f32x8 acc[2][4] = {};
    const __bf16* arow0 = X + (rowbase + n)      * E_;
    const __bf16* arow1 = X + (rowbase + 16 + n) * E_;
    const __bf16* wrow  = W + (colbase + n)      * E_;

    for (int kb = 0; kb < E_; kb += 32) {
        const int k0 = kb + 8 * hlf;
        const int k1 = kb + 16 + 8 * hlf;
        __builtin_prefetch(arow0 + kb + 128, 0, 1);
        bf16x16 a0 = load_frag(arow0, k0, k1);
        bf16x16 a1 = load_frag(arow1, k0, k1);
#pragma unroll
        for (int f = 0; f < 4; ++f) {
            bf16x16 b = load_frag(wrow + f * 16 * E_, k0, k1);
            acc[0][f] = wmma_bf16(a0, b, acc[0][f]);
            acc[1][f] = wmma_bf16(a1, b, acc[1][f]);
        }
    }

    // Scatter: col c -> head h = c/192, type t = (c%192)/64, d = c%64
#pragma unroll
    for (int i = 0; i < 2; ++i) {
#pragma unroll
        for (int f = 0; f < 4; ++f) {
#pragma unroll
            for (int r = 0; r < 8; ++r) {
                const int row = rowbase + 16 * i + r + 8 * hlf;
                const int c   = colbase + 16 * f + n;
                const int hh  = c / 192;
                const int rem = c - hh * 192;
                const int t   = rem >> 6;          // 0=q,1=k,2=v
                const int d   = rem & 63;
                const int b   = row >> 11;         // /S_
                const int s   = row & (S_ - 1);
                const __bf16 bv = (__bf16)acc[i][f][r];
                if (t == 0)       Qb[((b * H_ + hh) * S_ + s) * DH_ + d] = bv;
                else if (t == 1)  Kb[((b * H_ + hh) * S_ + s) * DH_ + d] = bv;
                else              Vt[((b * H_ + hh) * DH_ + d) * S_ + s] = bv;
            }
        }
    }
}

// ---------------------------------------------------------------------------
// Stage 3: flash attention.  One wave per (b,h, 16-query tile); 32-key blocks.
// scores: Q(16x64) x Kblk^T(64x32) -> two 16x16 C frags; online softmax with
// 16-lane butterflies; P re-laid out C->A via a private LDS slab; O += P x V.
// ---------------------------------------------------------------------------
__global__ __launch_bounds__(128) void attn_kernel(const unsigned short* __restrict__ Qus,
                                                   const unsigned short* __restrict__ Kus,
                                                   const unsigned short* __restrict__ Vtus,
                                                   unsigned short* __restrict__ AOus) {
    __shared__ __align__(16) __bf16 plds[4][16 * 32];   // 1 KB per wave

    const __bf16* Qb = reinterpret_cast<const __bf16*>(Qus);
    const __bf16* Kb = reinterpret_cast<const __bf16*>(Kus);
    const __bf16* Vt = reinterpret_cast<const __bf16*>(Vtus);
    __bf16* AO = reinterpret_cast<__bf16*>(AOus);

    const int lane = threadIdx.x & 31;
    const int wave = threadIdx.x >> 5;
    const int gid  = blockIdx.x * 4 + wave;     // 8192 tiles
    const int qt   = gid & 127;                 // S/16 query tiles
    const int bh   = gid >> 7;                  // b*H + h
    const int n    = lane & 15;
    const int hlf  = lane >> 4;
    const int qbase = qt * 16;

    // Q fragments (row = this lane's query), dh chunks 0..31 and 32..63
    const __bf16* Qp = Qb + (bh * S_ + qbase + n) * DH_;
    bf16x16 qf0 = load_frag(Qp, 0 + 8 * hlf, 16 + 8 * hlf);
    bf16x16 qf1 = load_frag(Qp, 32 + 8 * hlf, 48 + 8 * hlf);

    const __bf16* Kbase = Kb + bh * S_ * DH_;
    const __bf16* Vbase = Vt + bh * DH_ * S_;

    f32x8 o[4] = {};
    float m[8], l[8];
#pragma unroll
    for (int r = 0; r < 8; ++r) { m[r] = -1.0e30f; l[r] = 0.0f; }

    const float cs = 0.125f * 1.44269504088896f;   // (1/sqrt(dh)) * log2(e)

    for (int kb = 0; kb < S_; kb += 32) {
        // ----- scores for this 32-key block: two 16x16 C fragments -----
        const __bf16* kr0 = Kbase + (kb + n) * DH_;
        const __bf16* kr1 = Kbase + (kb + 16 + n) * DH_;
        __builtin_prefetch(kr0 + 32 * DH_, 0, 1);
        f32x8 s0 = {}, s1 = {};
        s0 = wmma_bf16(qf0, load_frag(kr0, 0 + 8 * hlf, 16 + 8 * hlf), s0);
        s0 = wmma_bf16(qf1, load_frag(kr0, 32 + 8 * hlf, 48 + 8 * hlf), s0);
        s1 = wmma_bf16(qf0, load_frag(kr1, 0 + 8 * hlf, 16 + 8 * hlf), s1);
        s1 = wmma_bf16(qf1, load_frag(kr1, 32 + 8 * hlf, 48 + 8 * hlf), s1);

        // ----- online softmax (per fragment row = r + 8*hlf) -----
#pragma unroll
        for (int r = 0; r < 8; ++r) {
            float a0 = s0[r] * cs;
            float a1 = s1[r] * cs;
            float bm = rmax16(fmaxf(a0, a1));
            float mn = fmaxf(m[r], bm);
            float alpha = exp2f(m[r] - mn);
            m[r] = mn;
            float p0 = exp2f(a0 - mn);
            float p1 = exp2f(a1 - mn);
            float ts = rsum16(p0 + p1);
            l[r] = l[r] * alpha + ts;
#pragma unroll
            for (int f = 0; f < 4; ++f) o[f][r] = o[f][r] * alpha;
            // stage P (bf16) to LDS in row-major 16x32 for C->A relayout
            plds[wave][(r + 8 * hlf) * 32 + n]      = (__bf16)p0;
            plds[wave][(r + 8 * hlf) * 32 + 16 + n] = (__bf16)p1;
        }

        // ----- read P back as a 16x32 A fragment (ds_load_b128 x2) -----
        bf16x16 pa = load_frag(&plds[wave][n * 32], 0 + 8 * hlf, 16 + 8 * hlf);

        // ----- O += P x Vblk (Vt rows are contiguous along keys) -----
#pragma unroll
        for (int f = 0; f < 4; ++f) {
            const __bf16* vr = Vbase + (16 * f + n) * S_;
            bf16x16 vb = load_frag(vr, kb + 8 * hlf, kb + 16 + 8 * hlf);
            o[f] = wmma_bf16(pa, vb, o[f]);
        }
    }

    // ----- finalize: O /= l, store to (B*S, E) row-major bf16 -----
    const int b  = bh >> 4;
    const int hh = bh & 15;
#pragma unroll
    for (int r = 0; r < 8; ++r) {
        const float inv = 1.0f / l[r];
        const int srow = qbase + r + 8 * hlf;
        const int base = (b * S_ + srow) * E_ + hh * DH_;
#pragma unroll
        for (int f = 0; f < 4; ++f)
            AO[base + 16 * f + n] = (__bf16)(o[f][r] * inv);
    }
}

// ---------------------------------------------------------------------------
// Stage 4: output GEMM.  out(8192x1024,f32) = AO(8192x1024,bf16) * Wo^T.
// ---------------------------------------------------------------------------
__global__ __launch_bounds__(128) void gemm_out(const unsigned short* __restrict__ Aus,
                                                const unsigned short* __restrict__ Wus,
                                                float* __restrict__ out) {
    const __bf16* A = reinterpret_cast<const __bf16*>(Aus);
    const __bf16* W = reinterpret_cast<const __bf16*>(Wus);

    const int lane = threadIdx.x & 31;
    const int wave = threadIdx.x >> 5;
    const int gid  = blockIdx.x * 4 + wave;     // 4096 wave tiles
    const int tn   = gid & 15;                  // 16 col tiles of 64 (N=1024)
    const int tm   = gid >> 4;                  // 256 row tiles of 32
    const int n    = lane & 15;
    const int hlf  = lane >> 4;
    const int rowbase = tm * 32;
    const int colbase = tn * 64;

    f32x8 acc[2][4] = {};
    const __bf16* arow0 = A + (rowbase + n)      * E_;
    const __bf16* arow1 = A + (rowbase + 16 + n) * E_;
    const __bf16* wrow  = W + (colbase + n)      * E_;

    for (int kb = 0; kb < E_; kb += 32) {
        const int k0 = kb + 8 * hlf;
        const int k1 = kb + 16 + 8 * hlf;
        __builtin_prefetch(wrow + kb + 128, 0, 1);
        bf16x16 a0 = load_frag(arow0, k0, k1);
        bf16x16 a1 = load_frag(arow1, k0, k1);
#pragma unroll
        for (int f = 0; f < 4; ++f) {
            bf16x16 b = load_frag(wrow + f * 16 * E_, k0, k1);
            acc[0][f] = wmma_bf16(a0, b, acc[0][f]);
            acc[1][f] = wmma_bf16(a1, b, acc[1][f]);
        }
    }

#pragma unroll
    for (int i = 0; i < 2; ++i)
#pragma unroll
        for (int f = 0; f < 4; ++f)
#pragma unroll
            for (int r = 0; r < 8; ++r)
                out[(rowbase + 16 * i + r + 8 * hlf) * E_ + colbase + 16 * f + n] = acc[i][f][r];
}

// ---------------------------------------------------------------------------
extern "C" void kernel_launch(void* const* d_in, const int* in_sizes, int n_in,
                              void* d_out, int out_size, void* d_ws, size_t ws_size,
                              hipStream_t stream) {
    (void)in_sizes; (void)n_in; (void)out_size; (void)ws_size;
    const float* x    = (const float*)d_in[0];   // (4,2048,1024) f32
    const float* wqkv = (const float*)d_in[1];   // (3072,1024)   f32
    const float* wo   = (const float*)d_in[2];   // (1024,1024)   f32
    float* out = (float*)d_out;                  // (4,2048,1024) f32

    char* ws = (char*)d_ws;
    size_t off = 0;
    auto alloc = [&](size_t bytes) -> unsigned short* {
        unsigned short* p = (unsigned short*)(ws + off);
        off += (bytes + 255) & ~(size_t)255;
        return p;
    };
    const size_t MS = (size_t)B_ * S_;                        // 8192 rows
    unsigned short* xb    = alloc(MS * E_ * 2);               // 16 MB
    unsigned short* wqkvb = alloc((size_t)F_ * E_ * 2);       //  6 MB
    unsigned short* wob   = alloc((size_t)E_ * E_ * 2);       //  2 MB
    unsigned short* Qb    = alloc(MS * E_ * 2);               // 16 MB (b,h,s,d)
    unsigned short* Kb    = alloc(MS * E_ * 2);               // 16 MB (b,h,s,d)
    unsigned short* Vt    = alloc(MS * E_ * 2);               // 16 MB (b,h,d,s)
    unsigned short* AO    = alloc(MS * E_ * 2);               // 16 MB (b,s,e)

    // Stage 1: conversions
    {
        int n4 = (int)(MS * E_ / 4);
        cvt_f32_bf16<<<(n4 + 255) / 256, 256, 0, stream>>>(x, xb, n4);
        n4 = F_ * E_ / 4;
        cvt_f32_bf16<<<(n4 + 255) / 256, 256, 0, stream>>>(wqkv, wqkvb, n4);
        n4 = E_ * E_ / 4;
        cvt_f32_bf16<<<(n4 + 255) / 256, 256, 0, stream>>>(wo, wob, n4);
    }
    // Stage 2: QKV projection (12288 wave tiles, 4 waves/block)
    gemm_qkv<<<12288 / 4, 128, 0, stream>>>(xb, wqkvb, Qb, Kb, Vt);
    // Stage 3: attention (8192 wave tiles)
    attn_kernel<<<8192 / 4, 128, 0, stream>>>(Qb, Kb, Vt, AO);
    // Stage 4: output projection (4096 wave tiles)
    gemm_out<<<4096 / 4, 128, 0, stream>>>(AO, wob, out);
}
